// MUSE_65601330479517
// MI455X (gfx1250) — matrix operand order repaired and checked by reading, not compiled, not run on hardware
//
#include <hip/hip_runtime.h>
#include <math.h>

// ---- problem constants (match reference) ----
constexpr int kB = 4096;
constexpr int kN = 200;
constexpr int kH = 128;

typedef __attribute__((ext_vector_type(16))) __bf16 v16bf;
typedef __attribute__((ext_vector_type(8)))  __bf16 v8bf;
typedef __attribute__((ext_vector_type(8)))  float  v8f;

static __device__ __forceinline__ float sigmoidf_(float x) {
  return 1.0f / (1.0f + __expf(-x));
}

// fp32 -> bf16 fragment conversion (packed v_cvt_pk_bf16_f32 on gfx1250)
static __device__ __forceinline__ v16bf cvt16(float4 a, float4 b, float4 c, float4 d) {
  v16bf r;
  r[0]  = (__bf16)a.x; r[1]  = (__bf16)a.y; r[2]  = (__bf16)a.z; r[3]  = (__bf16)a.w;
  r[4]  = (__bf16)b.x; r[5]  = (__bf16)b.y; r[6]  = (__bf16)b.z; r[7]  = (__bf16)b.w;
  r[8]  = (__bf16)c.x; r[9]  = (__bf16)c.y; r[10] = (__bf16)c.z; r[11] = (__bf16)c.w;
  r[12] = (__bf16)d.x; r[13] = (__bf16)d.y; r[14] = (__bf16)d.z; r[15] = (__bf16)d.w;
  return r;
}

// load 16 contiguous bf16 (32B) into a fragment: two b128 loads, no cvt
static __device__ __forceinline__ v16bf load16bf(const __bf16* p) {
  v8bf lo = *(const v8bf*)p;
  v8bf hi = *(const v8bf*)(p + 8);
  return __builtin_shufflevector(lo, hi, 0, 1, 2, 3, 4, 5, 6, 7,
                                 8, 9, 10, 11, 12, 13, 14, 15);
}

// ============================================================================
// GEMM with A-fragment reuse: C[M x Nout] = X[M x K](f32) * W[Nout x K](bf16)^T
// + bias.  One wave owns one 16-row m-tile and accumulates NT 16x16 tiles
// (n-tiles [blockIdx.y*NT, +NT)); X streamed once per n-group; W pre-converted
// bf16 so B fragments load directly into their 8 VGPRs.  sched_barrier caps
// in-flight B fragments -> bounded register pressure, no spills.
// Requires M%16==0, K%32==0 (true at all call sites).
// ============================================================================
template <int NT, bool OUT_BF16>
__global__ void gemm_xwt_bias(const float* __restrict__ X, int ldx,
                              const __bf16* __restrict__ W, int ldw,
                              const float* __restrict__ bias,
                              void* __restrict__ Cout, int ldc,
                              int M, int K) {
  const int lane = threadIdx.x & 31;
  const int wid  = blockIdx.x * (blockDim.x >> 5) + (threadIdx.x >> 5);  // m-tile
  if (wid >= (M >> 4)) return;             // wave-uniform exit, EXEC stays full
  const int n0t = blockIdx.y * NT;

  const int g  = lane >> 4;
  const int lm = lane & 15;
  const size_t arow = (size_t)(wid * 16 + lm) * (size_t)ldx;

  v8f acc[NT] = {};

  for (int kk = 0; kk < K; kk += 32) {
    const float* ap = X + arow + kk + 16 * g;
    v16bf Av = cvt16(*(const float4*)(ap),      *(const float4*)(ap + 4),
                     *(const float4*)(ap + 8),  *(const float4*)(ap + 12));
    #pragma unroll
    for (int t = 0; t < NT; ++t) {
      v16bf Bv = load16bf(W + (size_t)((n0t + t) * 16 + lm) * (size_t)ldw + kk + 16 * g);
      acc[t] = __builtin_amdgcn_wmma_f32_16x16x32_bf16(false, Av, false, Bv,
                                                       (short)0, acc[t], false, false);
      __builtin_amdgcn_sched_barrier(0);   // cap in-flight B fragments
    }
  }

  #pragma unroll
  for (int t = 0; t < NT; ++t) {
    const int ccol = (n0t + t) * 16 + lm;
    const float bv = bias[ccol];
    #pragma unroll
    for (int v = 0; v < 8; ++v) {
      int row = wid * 16 + v + 8 * g;
      if (OUT_BF16)
        ((__bf16*)Cout)[(size_t)row * (size_t)ldc + ccol] = (__bf16)(acc[t][v] + bv);
      else
        ((float*)Cout)[(size_t)row * (size_t)ldc + ccol] = acc[t][v] + bv;
    }
  }
}

// ============================================================================
// Batched adjacency GEMM: C[b](kN x kH) = Adj[b](kN x kN, f32) * Hm[b](kN x kH,
// bf16) + bias.  Adj row stride 2*kN (A_in/A_out slices), Hm bf16 row stride
// 2*kH, C f32 row stride 2*kH.  Wave owns one m-tile + all 8 n-tiles.
//
// Edge handling: out-of-range OUTPUT rows are masked at the store, so A values
// for m>=200 are don't-cares -> a row-CLAMPED pointer makes all A loads safe
// and every full k-step takes the fast b128 path for every m-tile.  Only the
// k-epilogue (kk=192; valid k = 192..199 -> fragment slots g==0,i<8) needs
// value masking: unconditional loads + one per-lane select, rest are zeros.
// ============================================================================
__global__ void adj_gemm(const float* __restrict__ Abase,
                         const __bf16* __restrict__ Hbase,
                         const float* __restrict__ bias,
                         float* __restrict__ Cbase) {
  constexpr int TM = (kN + 15) / 16;       // 13
  constexpr int NT = kH / 16;              // 8
  const int b    = blockIdx.y;
  const int lane = threadIdx.x & 31;
  const int wid  = blockIdx.x * (blockDim.x >> 5) + (threadIdx.x >> 5);  // m-tile
  if (wid >= TM) return;

  const float*  Aj = Abase + (size_t)b * kN * (2 * kN);
  const __bf16* Hm = Hbase + (size_t)b * kN * (2 * kH);
  float*        Cm = Cbase + (size_t)b * kN * (2 * kH);

  const int g  = lane >> 4;
  const int lm = lane & 15;
  const int am = wid * 16 + lm;
  // clamped row pointer: always-valid loads; clamped rows' values are unused
  const float* arow = Aj + (size_t)(am < kN ? am : kN - 1) * (2 * kN);

  v8f acc[NT] = {};

  // ---- main loop: 6 full k-steps (k = 0..191), fast path for all m-tiles ----
  for (int kk = 0; kk < 192; kk += 32) {
    const float* ap = arow + kk + 16 * g;
    v16bf Av = cvt16(*(const float4*)(ap),      *(const float4*)(ap + 4),
                     *(const float4*)(ap + 8),  *(const float4*)(ap + 12));
    #pragma unroll
    for (int t = 0; t < NT; ++t) {
      const int cn = t * 16 + lm;
      v16bf Bv;
      #pragma unroll
      for (int i = 0; i < 16; ++i)
        Bv[i] = Hm[(size_t)(kk + 16 * g + i) * (2 * kH) + cn];
      acc[t] = __builtin_amdgcn_wmma_f32_16x16x32_bf16(false, Av, false, Bv,
                                                       (short)0, acc[t], false, false);
      __builtin_amdgcn_sched_barrier(0);
    }
  }

  // ---- epilogue: kk = 192, valid k = 192..199 (slots g==0, i<8 only) ----
  {
    const bool sel = (g == 0);
    const __bf16 z = (__bf16)0.0f;
    float4 a0 = *(const float4*)(arow + 192);     // valid for all lanes
    float4 a1 = *(const float4*)(arow + 196);
    v16bf Av = {};
    Av[0] = sel ? (__bf16)a0.x : z; Av[1] = sel ? (__bf16)a0.y : z;
    Av[2] = sel ? (__bf16)a0.z : z; Av[3] = sel ? (__bf16)a0.w : z;
    Av[4] = sel ? (__bf16)a1.x : z; Av[5] = sel ? (__bf16)a1.y : z;
    Av[6] = sel ? (__bf16)a1.z : z; Av[7] = sel ? (__bf16)a1.w : z;
    #pragma unroll
    for (int t = 0; t < NT; ++t) {
      const int cn = t * 16 + lm;
      v16bf Bv = {};
      #pragma unroll
      for (int i = 0; i < 8; ++i) {
        __bf16 v = Hm[(size_t)(192 + i) * (2 * kH) + cn];   // rows 192..199 valid
        Bv[i] = sel ? v : z;
      }
      acc[t] = __builtin_amdgcn_wmma_f32_16x16x32_bf16(false, Av, false, Bv,
                                                       (short)0, acc[t], false, false);
      __builtin_amdgcn_sched_barrier(0);
    }
  }

  #pragma unroll
  for (int t = 0; t < NT; ++t) {
    const int cn = t * 16 + lm;
    const float bv = bias[cn];
    #pragma unroll
    for (int v = 0; v < 8; ++v) {
      int row = wid * 16 + v + 8 * g;
      if (row < kN) Cm[(size_t)row * (2 * kH) + cn] = acc[t][v] + bv;
    }
  }
}

// ============================================================================
// Elementwise / gather kernels
// ============================================================================
__global__ void to_bf16(const float* __restrict__ src, __bf16* __restrict__ dst, int n) {
  int i = blockIdx.x * blockDim.x + threadIdx.x;
  if (i < n) dst[i] = (__bf16)src[i];
}

__global__ void embed_gather(const int* __restrict__ items,
                             const float* __restrict__ emb,
                             float* __restrict__ hid) {
  size_t idx = (size_t)blockIdx.x * blockDim.x + threadIdx.x;   // B*N*(H/4)
  size_t row = idx >> 5;                 // H/4 == 32
  int c4 = (int)(idx & 31);
  if (row >= (size_t)kB * kN) return;
  int it = items[row];
  ((float4*)hid)[row * 32 + c4] = ((const float4*)emb)[(size_t)it * 32 + c4];
}

static __device__ __forceinline__ float gru1(float ir, float ii, float in_,
                                             float hr, float hi, float hn, float hv) {
  float r  = sigmoidf_(ir + hr);
  float z  = sigmoidf_(ii + hi);
  float ng = tanhf(in_ + r * hn);
  return ng + z * (hv - ng);
}

__global__ void gru_gates(const float* __restrict__ gi, const float* __restrict__ gh,
                          float* __restrict__ hid, size_t total4) {
  size_t idx = (size_t)blockIdx.x * blockDim.x + threadIdx.x;   // B*N*(H/4)
  if (idx >= total4) return;
  size_t m = idx >> 5;
  int h4 = (int)(idx & 31);
  const float4* g4i = (const float4*)gi;
  const float4* g4h = (const float4*)gh;
  size_t gb = m * 96 + h4;                // 3H floats = 96 float4
  float4 ir = g4i[gb], ii = g4i[gb + 32], in_ = g4i[gb + 64];
  float4 hr = g4h[gb], hi = g4h[gb + 32], hn  = g4h[gb + 64];
  float4 hv = ((const float4*)hid)[idx];
  float4 o;
  o.x = gru1(ir.x, ii.x, in_.x, hr.x, hi.x, hn.x, hv.x);
  o.y = gru1(ir.y, ii.y, in_.y, hr.y, hi.y, hn.y, hv.y);
  o.z = gru1(ir.z, ii.z, in_.z, hr.z, hi.z, hn.z, hv.z);
  o.w = gru1(ir.w, ii.w, in_.w, hr.w, hi.w, hn.w, hv.w);
  ((float4*)hid)[idx] = o;
}

__global__ void seq_gather(const float* __restrict__ hid, const int* __restrict__ alias,
                           float* __restrict__ out) {
  size_t idx = (size_t)blockIdx.x * blockDim.x + threadIdx.x;   // B*N*(H/4)
  size_t row = idx >> 5;
  int c4 = (int)(idx & 31);
  if (row >= (size_t)kB * kN) return;
  size_t b = row / kN;
  int a = alias[row];
  ((float4*)out)[row * 32 + c4] = ((const float4*)hid)[(b * kN + (size_t)a) * 32 + c4];
}

__global__ void ht_gather(const float* __restrict__ seqh, const int* __restrict__ lengths,
                          float* __restrict__ ht) {
  int idx = blockIdx.x * blockDim.x + threadIdx.x;              // B*(H/4)
  int b = idx >> 5, h4 = idx & 31;
  if (b >= kB) return;
  int li = lengths[b] - 1;
  li = li < 0 ? 0 : (li > kN - 1 ? kN - 1 : li);
  ((float4*)ht)[idx] = ((const float4*)seqh)[((size_t)b * kN + li) * 32 + h4];
}

__global__ void add_sig(const float* __restrict__ q1, const float* __restrict__ q2,
                        float* __restrict__ sig, size_t total4) {
  size_t idx = (size_t)blockIdx.x * blockDim.x + threadIdx.x;   // B*N*(H/4)
  if (idx >= total4) return;
  size_t row = idx >> 5;
  int h4 = (int)(idx & 31);
  size_t b = row / kN;
  float4 a = ((const float4*)q1)[b * 32 + h4];
  float4 c = ((const float4*)q2)[idx];
  float4 o;
  o.x = sigmoidf_(a.x + c.x); o.y = sigmoidf_(a.y + c.y);
  o.z = sigmoidf_(a.z + c.z); o.w = sigmoidf_(a.w + c.w);
  ((float4*)sig)[idx] = o;
}

__global__ void attn_reduce(const float* __restrict__ alp, const float* __restrict__ seqh,
                            const int* __restrict__ mask, const float* __restrict__ ht,
                            float* __restrict__ cat) {
  int idx = blockIdx.x * blockDim.x + threadIdx.x;              // B*(H/4)
  int b = idx >> 5, h4 = idx & 31;
  if (b >= kB) return;
  const float4* ap = (const float4*)alp  + (size_t)b * kN * 32 + h4;
  const float4* sp = (const float4*)seqh + (size_t)b * kN * 32 + h4;
  const int*    mp = mask + (size_t)b * kN;
  float4 acc = {0.f, 0.f, 0.f, 0.f};
  for (int l = 0; l < kN; ++l) {
    float4 a = ap[(size_t)l * 32];
    float4 s = sp[(size_t)l * 32];
    float  m = (float)mp[l];
    acc.x += a.x * s.x * m; acc.y += a.y * s.y * m;
    acc.z += a.z * s.z * m; acc.w += a.w * s.w * m;
  }
  ((float4*)cat)[(size_t)b * 64 + h4]      = acc;
  ((float4*)cat)[(size_t)b * 64 + 32 + h4] = ((const float4*)ht)[idx];
}

// ============================================================================
// host-side orchestration
// ============================================================================
extern "C" void kernel_launch(void* const* d_in, const int* in_sizes, int n_in,
                              void* d_out, int out_size, void* d_ws, size_t ws_size,
                              hipStream_t stream) {
  const float* A     = (const float*)d_in[0];
  const int*   items = (const int*)d_in[1];
  const int*   alias = (const int*)d_in[2];
  const int*   mask  = (const int*)d_in[3];
  const int*   lens  = (const int*)d_in[4];
  const float* emb   = (const float*)d_in[5];
  const float* w_ih  = (const float*)d_in[6];
  const float* w_hh  = (const float*)d_in[7];
  const float* b_ih  = (const float*)d_in[8];
  const float* b_hh  = (const float*)d_in[9];
  const float* b_iah = (const float*)d_in[10];
  const float* b_oah = (const float*)d_in[11];
  const float* W_in  = (const float*)d_in[12];
  const float* b_in  = (const float*)d_in[13];
  const float* W_out = (const float*)d_in[14];
  const float* b_out = (const float*)d_in[15];
  const float* W1    = (const float*)d_in[16];
  const float* b1    = (const float*)d_in[17];
  const float* W2    = (const float*)d_in[18];
  const float* b2    = (const float*)d_in[19];
  const float* W3    = (const float*)d_in[20];
  const float* b3    = (const float*)d_in[21];
  const float* Wt    = (const float*)d_in[22];
  const float* bt    = (const float*)d_in[23];

  const size_t BNH = (size_t)kB * kN * kH;     // 104,857,600
  const int    BN  = kB * kN;                  // 819,200

  // ---- workspace layout (byte-based; gi/gh live past 2GB) ----
  char* base    = (char*)d_ws;
  float*  hid    = (float*)(base);                     // BNH f32
  __bf16* h_io   = (__bf16*)(base + BNH * 4);          // 2*BNH bf16 (h_in|h_out)
  float*  inputs = (float*)(base + 2 * BNH * 4);       // 2*BNH f32 (in_in|in_out)
  float*  gi     = (float*)(base + 4 * BNH * 4);       // 3*BNH f32
  float*  gh     = (float*)(base + 7 * BNH * 4);       // 3*BNH f32
  __bf16* wbf    = (__bf16*)(base + 10 * BNH * 4);     // bf16 weights (~0.5MB)
  // readout reuse
  float* q2v  = gi;                                    // B*N*H
  float* sigv = gh;                                    // B*N*H
  float* alpv = inputs;                                // B*N*H
  float* htv  = (float*)(base + BNH * 4);              // B*H
  float* q1v  = htv + (size_t)kB * kH;                 // B*H
  float* catv = q1v + (size_t)kB * kH;                 // B*2H

  // ---- bf16 weight copies (tiny, converted on device each call) ----
  __bf16* w_ih_b  = wbf;                 // 384*256
  __bf16* w_hh_b  = w_ih_b  + 98304;     // 384*128
  __bf16* W_in_b  = w_hh_b  + 49152;     // 128*128
  __bf16* W_out_b = W_in_b  + 16384;
  __bf16* W1_b    = W_out_b + 16384;
  __bf16* W2_b    = W1_b    + 16384;
  __bf16* W3_b    = W2_b    + 16384;
  __bf16* Wt_b    = W3_b    + 16384;     // 128*256

  auto conv = [&](const float* s, __bf16* d, int n) {
    to_bf16<<<(n + 255) / 256, 256, 0, stream>>>(s, d, n);
  };
  conv(w_ih, w_ih_b, 98304);  conv(w_hh, w_hh_b, 49152);
  conv(W_in, W_in_b, 16384);  conv(W_out, W_out_b, 16384);
  conv(W1, W1_b, 16384);      conv(W2, W2_b, 16384);
  conv(W3, W3_b, 16384);      conv(Wt, Wt_b, 32768);

  // Nout == 128: one n-group of 8 tiles.  Nout == 384: three n-groups.
  auto gemm128 = [&](const float* X, int ldx, const __bf16* W, int ldw,
                     const float* bias, float* C, int ldc, int M, int K) {
    dim3 grid((M / 16 + 3) / 4, 1);
    gemm_xwt_bias<8, false><<<grid, 128, 0, stream>>>(X, ldx, W, ldw, bias, C, ldc, M, K);
  };
  auto gemm128bf = [&](const float* X, int ldx, const __bf16* W, int ldw,
                       const float* bias, __bf16* C, int ldc, int M, int K) {
    dim3 grid((M / 16 + 3) / 4, 1);
    gemm_xwt_bias<8, true><<<grid, 128, 0, stream>>>(X, ldx, W, ldw, bias, C, ldc, M, K);
  };
  auto gemm384 = [&](const float* X, int ldx, const __bf16* W, int ldw,
                     const float* bias, float* C, int ldc, int M, int K) {
    dim3 grid((M / 16 + 3) / 4, 3);
    gemm_xwt_bias<8, false><<<grid, 128, 0, stream>>>(X, ldx, W, ldw, bias, C, ldc, M, K);
  };

  float* outp = (float*)d_out;

  // hidden = emb_table[items]
  embed_gather<<<(unsigned)((BNH / 4 + 255) / 256), 256, 0, stream>>>(items, emb, hid);

  // ---- 2 GNN steps ----
  const dim3 adjGrid(4, kB);                   // 13 m-tiles / 4 waves per block
  for (int s = 0; s < 2; ++s) {
    gemm128bf(hid, kH, W_in_b,  kH, b_in,  h_io,      2 * kH, BN, kH);
    gemm128bf(hid, kH, W_out_b, kH, b_out, h_io + kH, 2 * kH, BN, kH);
    adj_gemm<<<adjGrid, 128, 0, stream>>>(A,      h_io,      b_iah, inputs);
    adj_gemm<<<adjGrid, 128, 0, stream>>>(A + kN, h_io + kH, b_oah, inputs + kH);
    gemm384(inputs, 2 * kH, w_ih_b, 2 * kH, b_ih, gi, 3 * kH, BN, 2 * kH);
    gemm384(hid,    kH,     w_hh_b, kH,     b_hh, gh, 3 * kH, BN, kH);
    gru_gates<<<(unsigned)((BNH / 4 + 255) / 256), 256, 0, stream>>>(gi, gh, hid, BNH / 4);
  }

  // ---- readout ----
  seq_gather<<<(unsigned)((BNH / 4 + 255) / 256), 256, 0, stream>>>(hid, alias, outp);
  ht_gather<<<(kB * kH / 4 + 255) / 256, 256, 0, stream>>>(outp, lens, htv);
  gemm128(htv,  kH, W1_b, kH, b1, q1v, kH, kB, kH);
  gemm128(outp, kH, W2_b, kH, b2, q2v, kH, BN, kH);
  add_sig<<<(unsigned)((BNH / 4 + 255) / 256), 256, 0, stream>>>(q1v, q2v, sigv, BNH / 4);
  gemm128(sigv, kH, W3_b, kH, b3, alpv, kH, BN, kH);
  attn_reduce<<<(kB * kH / 4 + 255) / 256, 256, 0, stream>>>(alpv, outp, mask, htv, catv);
  gemm128(catv, 2 * kH, Wt_b, 2 * kH, bt, outp + BNH, kH, kB, 2 * kH);
}
